// DiConvNeXt_66786741453315
// MI455X (gfx1250) — compile-verified, matrix operand order checked by reading, not run on hardware
//
#include <hip/hip_runtime.h>
#include <hip/hip_bf16.h>
#include <math.h>

typedef __bf16 bf16_t;
typedef __attribute__((ext_vector_type(16))) __bf16 v16bf;
typedef __attribute__((ext_vector_type(8)))  __bf16 v8bf;
typedef __attribute__((ext_vector_type(8)))  float  v8f;

#define S_SPATIAL (64*64*64)
#define DIM 96
#define HID 384

// Branch-free exact-style GELU: erf via Abramowitz-Stegun 7.1.26 (|err|<1.5e-7)
__device__ __forceinline__ float gelu_f(float v) {
  float xs  = v * 0.70710678118654752f;      // x / sqrt(2)
  float axs = fabsf(xs);
  float t   = __builtin_amdgcn_rcpf(1.0f + 0.3275911f * axs);
  float poly = t*(0.254829592f + t*(-0.284496736f +
               t*(1.421413741f + t*(-1.453152027f + t*1.061405429f))));
  float e = __expf(-axs*axs);                // v_exp_f32, branch-free
  float erfv = copysignf(1.0f - poly*e, xs); // v_bfi
  return 0.5f * v * (1.0f + erfv);
}

// ---------------------------------------------------------------------------
// Kernel A: depthwise 7x7x7 conv + bias (NCDHW) -> bf16 workspace (NCS).
// One block = one (batch, channel, 8x8x8 tile); each thread does an x-pair.
// ---------------------------------------------------------------------------
__global__ __launch_bounds__(256) void dwconv7_kernel(
    const float* __restrict__ x, const float* __restrict__ conv_w,
    const float* __restrict__ conv_b, bf16_t* __restrict__ yws)
{
  __shared__ float halo[14*14*14];
  int bc   = blockIdx.x >> 9;          // 512 tiles per (b,c)
  int tile = blockIdx.x & 511;
  int bb = bc / DIM, c = bc % DIM;
  int tz = (tile >> 6) & 7, ty = (tile >> 3) & 7, tx = tile & 7;
  int z0 = tz*8 - 3, y0 = ty*8 - 3, x0 = tx*8 - 3;
  const float* xc = x + ((size_t)(bb*DIM + c)) * S_SPATIAL;
  int t = threadIdx.x;
  for (int i = t; i < 14*14*14; i += 256) {
    int lz = i / 196, rem = i % 196, ly = rem / 14, lx = rem % 14;
    int gz = z0 + lz, gy = y0 + ly, gx = x0 + lx;
    float v = 0.f;
    if ((unsigned)gz < 64u && (unsigned)gy < 64u && (unsigned)gx < 64u)
      v = xc[(gz << 12) + (gy << 6) + gx];
    halo[i] = v;
  }
  __syncthreads();
  const float* wch = conv_w + c*343;   // uniform -> scalar loads
  float bias = conv_b[c];

  int o  = t << 1;                      // adjacent output pair along x
  int oz = o >> 6, oy = (o >> 3) & 7, ox = o & 7;   // ox even
  float acc0 = bias, acc1 = bias;
  for (int dz = 0; dz < 7; ++dz)
    for (int dy = 0; dy < 7; ++dy) {
      const float* hrow = &halo[(oz+dz)*196 + (oy+dy)*14 + ox];
      const float* wrow = &wch[(dz*7 + dy)*7];
      float hh[8];
      #pragma unroll
      for (int q = 0; q < 8; ++q) hh[q] = hrow[q];  // 8 loads serve 14 FMAs
      #pragma unroll
      for (int dx = 0; dx < 7; ++dx) {
        float w = wrow[dx];
        acc0 = fmaf(hh[dx],   w, acc0);
        acc1 = fmaf(hh[dx+1], w, acc1);
      }
    }
  int s = ((tz*8+oz) << 12) + ((ty*8+oy) << 6) + (tx*8+ox);
  size_t gbase = ((size_t)(bb*DIM + c))*S_SPATIAL + s;
  yws[gbase]   = (bf16_t)acc0;          // pair -> coalesced b32 store
  yws[gbase+1] = (bf16_t)acc1;
}

// ---------------------------------------------------------------------------
// Kernel B: fused LayerNorm -> GEMM1(bf16 WMMA) -> GELU -> GEMM2 -> scale+res
// 128 spatial rows per block; 8 waves = 8 M-tiles of 16.
// ---------------------------------------------------------------------------
#define A_STRIDE  104   // 96 + pad, 8*13 -> bank-coprime b128 rows
#define H_STRIDE  392   // 384 + pad, 8*49
#define W1_STRIDE 104
#define W2_STRIDE 392
#define O_STRIDE  100

#define A_OFF     0
#define H_OFF     (128*A_STRIDE*2)                 // 26624
#define W1_OFF    (H_OFF + 128*H_STRIDE*2)         // 126976
#define W2_OFF    (W1_OFF + 384*W1_STRIDE*2)       // 206848
#define STATS_OFF (W2_OFF + 96*W2_STRIDE*2)        // 282112
#define SMEM_BYTES (STATS_OFF + 512*4)             // 284160 (< 320KB/WGP)
#define O_OFF     W1_OFF                            // reuse w1T space

// 16-bit WMMA fragment (A rows=M or B rows=N) from LDS, documented layout:
// lane half h reads K runs [k+8h .. k+8h+7] and the same +16.
__device__ __forceinline__ v16bf load_frag(const bf16_t* base, int stride,
                                           int row, int kbase, int lane) {
  int half = lane >> 4;
  const bf16_t* p = base + row*stride + kbase + half*8;
  v8bf lo = *(const v8bf*)p;
  v8bf hi = *(const v8bf*)(p + 16);
  v16bf r;
  #pragma unroll
  for (int e = 0; e < 8; ++e) { r[e] = lo[e]; r[e+8] = hi[e]; }
  return r;
}

__global__ __launch_bounds__(256) void lnmlp_kernel(
    const bf16_t* __restrict__ yws, const float* __restrict__ x,
    const float* __restrict__ ln_g, const float* __restrict__ ln_b,
    const float* __restrict__ w1p, const float* __restrict__ b1p,
    const float* __restrict__ w2p, const float* __restrict__ b2p,
    const float* __restrict__ scale, float* __restrict__ out)
{
  extern __shared__ char smem[];
  bf16_t* Ald   = (bf16_t*)(smem + A_OFF);
  bf16_t* Hld   = (bf16_t*)(smem + H_OFF);
  bf16_t* W1T   = (bf16_t*)(smem + W1_OFF);
  bf16_t* W2T   = (bf16_t*)(smem + W2_OFF);
  float*  stats = (float* )(smem + STATS_OFF);
  float*  Old   = (float* )(smem + O_OFF);

  int t = threadIdx.x;
  int lane = t & 31, ln15 = lane & 15, half = lane >> 4;
  int wv = t >> 5;                      // wave id == M-tile id
  int bb = blockIdx.x >> 11;            // 2048 blocks per batch
  int s_base = (blockIdx.x & 2047) << 7;

  // Stage w1^T and w2^T as bf16 (coalesced global reads, transposed LDS write)
  for (int i = t; i < DIM*HID; i += 256) {
    int c = i / HID, e = i % HID;                 // w1[c][e]
    W1T[e*W1_STRIDE + c] = (bf16_t)w1p[i];
  }
  for (int i = t; i < HID*DIM; i += 256) {
    int e = i / DIM, c = i % DIM;                 // w2[e][c]
    W2T[c*W2_STRIDE + e] = (bf16_t)w2p[i];
  }

  // LayerNorm over channels: thread owns row r = t&127, channel parity h
  int r = t & 127, h = t >> 7;
  float val[48];
  float s1 = 0.f, s2 = 0.f;
  #pragma unroll
  for (int j = 0; j < 48; ++j) {
    int c = 2*j + h;
    float v = (float)yws[((size_t)(bb*DIM + c))*S_SPATIAL + s_base + r];
    val[j] = v; s1 += v; s2 += v*v;
  }
  stats[h*128 + r]       = s1;
  stats[256 + h*128 + r] = s2;
  __syncthreads();
  float S1 = stats[r] + stats[128 + r];
  float S2 = stats[256 + r] + stats[384 + r];
  float mu  = S1 * (1.f/96.f);
  float var = S2 * (1.f/96.f) - mu*mu;
  float rs  = rsqrtf(var + 1e-6f);
  #pragma unroll
  for (int j = 0; j < 48; ++j) {
    int c = 2*j + h;
    float a = (val[j] - mu) * rs * ln_g[c] + ln_b[c];
    Ald[r*A_STRIDE + c] = (bf16_t)a;
  }
  __syncthreads();

  // ---- GEMM1: A[128x96] x w1[96x384] + b1, GELU, -> H (bf16 in LDS)
  // unroll 2: two independent WMMA chains let the scheduler fill the
  // WMMA->VALU RAW hazard slots (4 coexec NOPs) with the other tile's GELU.
  v16bf a0 = load_frag(Ald, A_STRIDE, wv*16 + ln15, 0,  lane);
  v16bf a1 = load_frag(Ald, A_STRIDE, wv*16 + ln15, 32, lane);
  v16bf a2 = load_frag(Ald, A_STRIDE, wv*16 + ln15, 64, lane);
  #pragma unroll 2
  for (int n = 0; n < 24; ++n) {
    int ncol = n*16 + ln15;
    float bias = b1p[ncol];
    v16bf bf0 = load_frag(W1T, W1_STRIDE, ncol, 0,  lane);
    v16bf bf1 = load_frag(W1T, W1_STRIDE, ncol, 32, lane);
    v16bf bf2 = load_frag(W1T, W1_STRIDE, ncol, 64, lane);
    v8f acc = {};
    acc = __builtin_amdgcn_wmma_f32_16x16x32_bf16(false, a0, false, bf0, (short)0, acc, false, false);
    acc = __builtin_amdgcn_wmma_f32_16x16x32_bf16(false, a1, false, bf1, (short)0, acc, false, false);
    acc = __builtin_amdgcn_wmma_f32_16x16x32_bf16(false, a2, false, bf2, (short)0, acc, false, false);
    #pragma unroll
    for (int j = 0; j < 8; ++j) {
      float g = gelu_f(acc[j] + bias);
      int rr = wv*16 + j + 8*half;
      Hld[rr*H_STRIDE + ncol] = (bf16_t)g;
    }
  }
  __syncthreads();   // all waves done with W1T before O-tile reuses its space

  // ---- GEMM2: H[128x384] x w2[384x96] -> O (f32 in LDS)
  v8f acc2[6];
  #pragma unroll
  for (int n = 0; n < 6; ++n) { v8f z = {}; acc2[n] = z; }
  for (int k = 0; k < 12; ++k) {
    int kb = k*32;
    v16bf af = load_frag(Hld, H_STRIDE, wv*16 + ln15, kb, lane);
    v16bf bfr[6];
    #pragma unroll
    for (int n = 0; n < 6; ++n)
      bfr[n] = load_frag(W2T, W2_STRIDE, n*16 + ln15, kb, lane);
    #pragma unroll
    for (int n = 0; n < 6; ++n)
      acc2[n] = __builtin_amdgcn_wmma_f32_16x16x32_bf16(false, af, false, bfr[n], (short)0, acc2[n], false, false);
  }
  #pragma unroll
  for (int n = 0; n < 6; ++n) {
    int c = n*16 + ln15;
    #pragma unroll
    for (int j = 0; j < 8; ++j) {
      int rr = wv*16 + j + 8*half;
      Old[rr*O_STRIDE + c] = acc2[n][j];
    }
  }
  __syncthreads();

  // ---- epilogue: out = scale*(O + b2) + x, coalesced over spatial rows
  #pragma unroll
  for (int j = 0; j < 48; ++j) {
    int c = 2*j + h;
    size_t gi = ((size_t)(bb*DIM + c))*S_SPATIAL + s_base + r;
    out[gi] = scale[c] * (Old[r*O_STRIDE + c] + b2p[c]) + x[gi];
  }
}

// ---------------------------------------------------------------------------
extern "C" void kernel_launch(void* const* d_in, const int* in_sizes, int n_in,
                              void* d_out, int out_size, void* d_ws, size_t ws_size,
                              hipStream_t stream) {
  const float* x      = (const float*)d_in[0];
  const float* conv_w = (const float*)d_in[1];
  const float* conv_b = (const float*)d_in[2];
  const float* ln_g   = (const float*)d_in[3];
  const float* ln_b   = (const float*)d_in[4];
  const float* w1     = (const float*)d_in[5];
  const float* b1     = (const float*)d_in[6];
  const float* w2     = (const float*)d_in[7];
  const float* b2     = (const float*)d_in[8];
  const float* scale  = (const float*)d_in[9];
  float* out = (float*)d_out;
  bf16_t* yws = (bf16_t*)d_ws;   // 2*96*64^3 bf16 = ~100.7 MB

  // Kernel A: 2 batches * 96 channels * 512 tiles
  dwconv7_kernel<<<2*DIM*512, 256, 0, stream>>>(x, conv_w, conv_b, yws);

  // Kernel B: 524288 rows / 128 = 4096 blocks, ~278 KB dynamic LDS
  lnmlp_kernel<<<4096, 256, SMEM_BYTES, stream>>>(
      yws, x, ln_g, ln_b, w1, b1, w2, b2, scale, out);
}
